// RNNCell_11450382811374
// MI455X (gfx1250) — matrix operand order repaired and checked by reading, not compiled, not run on hardware
//
#include <hip/hip_runtime.h>
#include <hip/hip_bf16.h>
#include <math.h>

typedef __attribute__((ext_vector_type(16))) __bf16 v16bf;
typedef __attribute__((ext_vector_type(8)))  float  v8f;

#define RNN_N 64
#define RNN_L 512
#define RNN_H 1024
#define SCAN_WGS 64

// Build a 16-element bf16 fragment from 16 f32 values (4x float4).
__device__ __forceinline__ v16bf cvt16(float4 a, float4 b, float4 c, float4 d) {
    v16bf r;
    r[0]  = (__bf16)a.x; r[1]  = (__bf16)a.y; r[2]  = (__bf16)a.z; r[3]  = (__bf16)a.w;
    r[4]  = (__bf16)b.x; r[5]  = (__bf16)b.y; r[6]  = (__bf16)b.z; r[7]  = (__bf16)b.w;
    r[8]  = (__bf16)c.x; r[9]  = (__bf16)c.y; r[10] = (__bf16)c.z; r[11] = (__bf16)c.w;
    r[12] = (__bf16)d.x; r[13] = (__bf16)d.y; r[14] = (__bf16)d.z; r[15] = (__bf16)d.w;
    return r;
}

// ---------------------------------------------------------------------------
// Init: zero the grid-barrier counter and both bf16 h ping-pong buffers.
// ---------------------------------------------------------------------------
__global__ __launch_bounds__(256) void rnn_init_kernel(unsigned int* __restrict__ barrier,
                                                       unsigned int* __restrict__ hbuf_u32) {
    const int idx = blockIdx.x * blockDim.x + threadIdx.x;
    const int total = 2 * RNN_N * RNN_H / 2;  // 2 buffers, bf16 pairs as u32
    if (idx < total) hbuf_u32[idx] = 0u;
    if (idx == 0) *barrier = 0u;
}

// ---------------------------------------------------------------------------
// Phase 1: xp = x @ W_xh^T + b_xh, written into d_out (in-place consumed later).
// Each wave: one 16-row M-tile x four 16-col N-tiles, bf16 WMMA, f32 accum.
// Memory-roofline bound: ~268 MB HBM traffic; W_xh (4 MB) stays in L2.
// ---------------------------------------------------------------------------
__global__ __launch_bounds__(256) void rnn_xproj_kernel(const float* __restrict__ x,
                                                        const float* __restrict__ Wxh,
                                                        const float* __restrict__ bxh,
                                                        float* __restrict__ out) {
    const int lane  = threadIdx.x & 31;
    const int wave  = threadIdx.x >> 5;
    const int mtile = blockIdx.x * 8 + wave;      // 0..2047 (M = N*L = 32768)
    const int cbase = blockIdx.y * 64;            // 16 N-groups of 64 columns
    const int g     = lane >> 4;                  // half-wave group
    const int nr    = lane & 15;                  // A row / B column within tile

    const float* xrow = x + (size_t)(mtile * 16 + nr) * RNN_H;

    v8f acc0 = {}, acc1 = {}, acc2 = {}, acc3 = {};

    for (int kt = 0; kt < RNN_H / 32; ++kt) {
        // A fragment: K octets [g*8, g*8+8) and [16+g*8, 16+g*8+8) of this k-tile
        const int ka = kt * 32 + g * 8;
        float4 a0 = *(const float4*)(xrow + ka);
        float4 a1 = *(const float4*)(xrow + ka + 4);
        float4 a2 = *(const float4*)(xrow + ka + 16);
        float4 a3 = *(const float4*)(xrow + ka + 20);
        v16bf A = cvt16(a0, a1, a2, a3);

        // B fragments: B[k][n] = Wxh[c][k]; K octets [g*8, g*8+8) and [16+g*8, 16+g*8+8)
        const int kb = kt * 32 + g * 8;
#pragma unroll
        for (int j = 0; j < 4; ++j) {
            const float* wrow = Wxh + (size_t)(cbase + j * 16 + nr) * RNN_H + kb;
            float4 b0 = *(const float4*)(wrow);
            float4 b1 = *(const float4*)(wrow + 4);
            float4 b2 = *(const float4*)(wrow + 16);
            float4 b3 = *(const float4*)(wrow + 20);
            v16bf B = cvt16(b0, b1, b2, b3);
            if (j == 0) acc0 = __builtin_amdgcn_wmma_f32_16x16x32_bf16(false, A, false, B, (short)0, acc0, false, false);
            if (j == 1) acc1 = __builtin_amdgcn_wmma_f32_16x16x32_bf16(false, A, false, B, (short)0, acc1, false, false);
            if (j == 2) acc2 = __builtin_amdgcn_wmma_f32_16x16x32_bf16(false, A, false, B, (short)0, acc2, false, false);
            if (j == 3) acc3 = __builtin_amdgcn_wmma_f32_16x16x32_bf16(false, A, false, B, (short)0, acc3, false, false);
        }
    }

    // Epilogue: C layout -> row = mtile*16 + v + g*8, col = cbase + j*16 + nr
#pragma unroll
    for (int v = 0; v < 8; ++v) {
        const size_t r = (size_t)(mtile * 16 + v + g * 8);
        {
            int c = cbase + 0 * 16 + nr;
            out[r * RNN_H + c] = acc0[v] + bxh[c];
        }
        {
            int c = cbase + 1 * 16 + nr;
            out[r * RNN_H + c] = acc1[v] + bxh[c];
        }
        {
            int c = cbase + 2 * 16 + nr;
            out[r * RNN_H + c] = acc2[v] + bxh[c];
        }
        {
            int c = cbase + 3 * 16 + nr;
            out[r * RNN_H + c] = acc3[v] + bxh[c];
        }
    }
}

// ---------------------------------------------------------------------------
// Phase 2: persistent scan. 64 WGs; WG owns 16 columns; W_hh slice pre-packed
// into per-wave bf16 register fragments (zero weight traffic in the loop);
// grid barrier between steps. Dual accumulators halve the serial WMMA chain.
// ---------------------------------------------------------------------------
__global__ __launch_bounds__(256) void rnn_scan_kernel(const float* __restrict__ Whh,
                                                       const float* __restrict__ bhh,
                                                       float* __restrict__ out,   // xp in, h_t out (in place)
                                                       __bf16* __restrict__ hbuf, // [2][64*1024]
                                                       unsigned int* __restrict__ barrier) {
    __shared__ float partial[2][RNN_N][16];

    const int lane = threadIdx.x & 31;
    const int wave = threadIdx.x >> 5;
    const int m    = wave & 3;    // M-tile (batch rows m*16..m*16+15)
    const int kh   = wave >> 2;   // K half (kh*512)
    const int c0   = blockIdx.x * 16;
    const int g    = lane >> 4;
    const int nr   = lane & 15;

    // Preload B fragments for this wave's K-half: B[k][n] = Whh[c0+n][k].
    v16bf Bf[16];
    const float* wrow = Whh + (size_t)(c0 + nr) * RNN_H + kh * 512;
#pragma unroll
    for (int kt = 0; kt < 16; ++kt) {
        const float* p = wrow + kt * 32 + g * 8;
        float4 b0 = *(const float4*)(p);
        float4 b1 = *(const float4*)(p + 4);
        float4 b2 = *(const float4*)(p + 16);
        float4 b3 = *(const float4*)(p + 20);
        Bf[kt] = cvt16(b0, b1, b2, b3);
    }

    const int arow = m * 16 + nr;  // A row for this lane

    for (int t = 0; t < RNN_L; ++t) {
        const __bf16* hprev = hbuf + (size_t)(t & 1) * (RNN_N * RNN_H);
        __bf16*       hcur  = hbuf + (size_t)((t + 1) & 1) * (RNN_N * RNN_H);
        const __bf16* ap    = hprev + (size_t)arow * RNN_H + kh * 512;

        // Two independent accumulator chains (even/odd k-tiles) so back-to-back
        // WMMAs are not serialized on the same D registers.
        v8f acc_e = {}, acc_o = {};
#pragma unroll
        for (int kt = 0; kt < 16; kt += 2) {
            union { v16bf v; uint4 u[2]; } Ae, Ao;
            Ae.u[0] = *(const uint4*)(ap + kt * 32 + g * 8);
            Ae.u[1] = *(const uint4*)(ap + kt * 32 + 16 + g * 8);
            Ao.u[0] = *(const uint4*)(ap + (kt + 1) * 32 + g * 8);
            Ao.u[1] = *(const uint4*)(ap + (kt + 1) * 32 + 16 + g * 8);
            acc_e = __builtin_amdgcn_wmma_f32_16x16x32_bf16(false, Ae.v, false, Bf[kt],
                                                            (short)0, acc_e, false, false);
            acc_o = __builtin_amdgcn_wmma_f32_16x16x32_bf16(false, Ao.v, false, Bf[kt + 1],
                                                            (short)0, acc_o, false, false);
        }

        // Stash partial 16x16 f32 tile: (row_in_tile = v + g*8, col = nr);
        // merging the two accumulator chains here is free.
#pragma unroll
        for (int v = 0; v < 8; ++v)
            partial[kh][m * 16 + v + g * 8][nr] = acc_e[v] + acc_o[v];
        __syncthreads();

        // Fused epilogue: reduce K-halves, add bias + xp, tanh, emit f32 + bf16.
#pragma unroll
        for (int i = 0; i < 4; ++i) {
            const int flat = threadIdx.x + i * 256;   // 0..1023 over 64x16 tile
            const int r  = flat >> 4;                 // batch row
            const int cc = flat & 15;
            const int c  = c0 + cc;
            const float s = partial[0][r][cc] + partial[1][r][cc];
            const size_t oidx = ((size_t)r * RNN_L + t) * RNN_H + c;
            const float y = tanhf(s + bhh[c] + out[oidx]);
            out[oidx] = y;
            hcur[(size_t)r * RNN_H + c] = (__bf16)y;
        }
        __syncthreads();

        // Grid-wide barrier (monotonic counter; all 64 WGs co-resident).
        if (threadIdx.x == 0) {
            __threadfence();
            atomicAdd(barrier, 1u);
            const unsigned int target = (unsigned int)SCAN_WGS * (unsigned int)(t + 1);
            while (*(volatile unsigned int*)barrier < target) {
                __builtin_amdgcn_s_sleep(1);
            }
            __threadfence();
        }
        __syncthreads();
    }
}

// ---------------------------------------------------------------------------
extern "C" void kernel_launch(void* const* d_in, const int* in_sizes, int n_in,
                              void* d_out, int out_size, void* d_ws, size_t ws_size,
                              hipStream_t stream) {
    const float* x    = (const float*)d_in[0];
    const float* W_hh = (const float*)d_in[1];
    const float* b_hh = (const float*)d_in[2];
    const float* W_xh = (const float*)d_in[3];
    const float* b_xh = (const float*)d_in[4];
    float* out = (float*)d_out;

    unsigned int* barrier = (unsigned int*)d_ws;
    __bf16* hbuf = (__bf16*)((char*)d_ws + 1024);

    // Zero barrier + both h ping-pong buffers (h_0 = 0).
    rnn_init_kernel<<<256, 256, 0, stream>>>(barrier, (unsigned int*)hbuf);

    // Phase 1: xp -> d_out. Grid: 2048 M-tiles / 8 waves, 16 column groups.
    rnn_xproj_kernel<<<dim3(256, 16), 256, 0, stream>>>(x, W_xh, b_xh, out);

    // Phase 2: persistent scan over 512 steps, in-place xp -> h in d_out.
    rnn_scan_kernel<<<SCAN_WGS, 256, 0, stream>>>(W_hh, b_hh, out, hbuf, barrier);
}
